// SSMBackbone_4312147165290
// MI455X (gfx1250) — matrix operand order repaired and checked by reading, not compiled, not run on hardware
//
#include <hip/hip_runtime.h>
#include <hip/hip_bf16.h>
#include <math.h>

// ---------------------------------------------------------------------------
// Types for CDNA5 WMMA (wave32): D = A(16x32 bf16) x B(32x16 bf16) + C(16x16 f32)
// ---------------------------------------------------------------------------
typedef __bf16 bf16;
typedef __attribute__((ext_vector_type(16))) __bf16 bf16x16;
typedef __attribute__((ext_vector_type(8)))  float  f32x8;
typedef __attribute__((ext_vector_type(4)))  unsigned int u32x4;
typedef __attribute__((ext_vector_type(8)))  int i32x8;
typedef __attribute__((ext_vector_type(4)))  int i32x4;

#define DEV __device__ __forceinline__

#if __has_builtin(__builtin_amdgcn_tensor_load_to_lds) && __has_builtin(__builtin_amdgcn_s_wait_tensorcnt)
#define USE_TDM 1
#else
#define USE_TDM 0
#endif

static constexpr int BB   = 512;           // batch
static constexpr int NT   = 256;           // tokens
static constexpr int DMOD = 128;           // model dim
static constexpr int DIN  = 256;           // inner dim
static constexpr int NLAY = 4;
static constexpr int MR   = BB * NT;       // 131072 rows

// ---------------------------------------------------------------------------
// WMMA fragment loaders: contiguous k order (k = 16*half + s) in both A and B
// ---------------------------------------------------------------------------

// A: 16x32 bf16 tile, A row-major with leading dim lda.
// lane<16 holds row M=lane, K in {0..15}; lane>=16 same row, K +16.
DEV bf16x16 load_a_frag(const bf16* __restrict__ A, int lda, size_t mBase, int kBase, int lane) {
  int half = lane >> 4;
  size_t m = mBase + (lane & 15);
  const bf16* row = A + m * (size_t)lda + kBase;
  bf16x16 a;
#pragma unroll
  for (int v = 0; v < 8; ++v) {
    int k = 2 * v + 16 * half;
    a[2 * v]     = row[k];
    a[2 * v + 1] = row[k + 1];
  }
  return a;
}

// B fragment where mathematical B = W^T, W is (N,K) row-major:
// B[k][n] = W[n][k]; lane col n = lane&15, K half selected by lane>>4.
DEV bf16x16 load_b_frag_wt(const bf16* __restrict__ W, int ldw, int nBase, int kBase, int lane) {
  int n  = nBase + (lane & 15);
  int kb = kBase + (lane >> 4) * 16;
  const bf16* row = W + (size_t)n * ldw + kb;
  bf16x16 b;
#pragma unroll
  for (int v = 0; v < 8; ++v) {
    b[2 * v]     = row[2 * v];
    b[2 * v + 1] = row[2 * v + 1];
  }
  return b;
}

// B fragment from K-pair-interleaved storage: element (k,n) lives at
// (k>>1)*512 + 2n + (k&1). Pair (k,k+1) is one contiguous 32-bit load.
DEV bf16x16 load_b_frag_il(const bf16* __restrict__ Bm, int nBase, int kBase, int lane) {
  int n  = nBase + (lane & 15);
  int kh = (kBase >> 1) + (lane >> 4) * 8;    // 16 K per half -> 8 interleaved rows
  const bf16* p = Bm + (size_t)kh * 512 + 2 * n;
  bf16x16 b;
#pragma unroll
  for (int v = 0; v < 8; ++v) {
    b[2 * v]     = p[(size_t)v * 512];
    b[2 * v + 1] = p[(size_t)v * 512 + 1];
  }
  return b;
}

DEV f32x8 wmma_bf16(bf16x16 a, bf16x16 b, f32x8 c) {
  return __builtin_amdgcn_wmma_f32_16x16x32_bf16(false, a, false, b, (short)0, c, false, false);
}

DEV float wave_sum32(float v) {
#pragma unroll
  for (int off = 16; off; off >>= 1) v += __shfl_xor(v, off, 32);
  return v;
}

// ---------------------------------------------------------------------------
// Elementwise / small kernels
// ---------------------------------------------------------------------------

__global__ void add_pos_kernel(const float* __restrict__ tok, const float* __restrict__ pos,
                               float* __restrict__ x) {
  size_t i = (size_t)blockIdx.x * 256 + threadIdx.x;        // < MR*128
  x[i] = tok[i] + pos[i % (size_t)(NT * DMOD)];
}

__global__ void cvt_bf16_kernel(const float* __restrict__ src, bf16* __restrict__ dst, int n) {
  int i = blockIdx.x * 256 + threadIdx.x;
  if (i < n) dst[i] = (bf16)src[i];
}

// Kk[layer][l] = sum_n C[n]*dB[n]*dA[n]^l with nan_to_num
__global__ void kk_kernel(const float* __restrict__ ssmB, const float* __restrict__ ssmC,
                          const float* __restrict__ ldt, float* __restrict__ Kk) {
  int layer = blockIdx.x;
  int l = threadIdx.x;
  float dt = __expf(ldt[layer]);
  dt = fminf(fmaxf(dt, 1e-4f), 1.0f);
  float acc = 0.f;
#pragma unroll
  for (int n = 0; n < 16; ++n) {
    float A  = -(float)(n + 1);
    float da = __expf(fminf(fmaxf(dt * A, -10.f), 10.f));
    float db = (da - 1.f) / fminf(A, -1e-4f) * ssmB[layer * 16 + n];
    float p  = __expf((float)l * __logf(da));               // da^l, da in (e^-10,1)
    acc += ssmC[layer * 16 + n] * db * p;
  }
  if (!__builtin_isfinite(acc)) acc = 0.f;
  Kk[layer * 256 + l] = acc;
}

// Lower-triangular Toeplitz, REVERSED kernel (conv is correlation with left pad):
// T[i][j] = Kk[255 - (i-j)] for i>=j else 0.
__global__ void tbuild_kernel(const float* __restrict__ Kk, bf16* __restrict__ T) {
  int i = blockIdx.x * 256 + threadIdx.x;                   // < 4*65536
  int layer = i >> 16;
  int r = (i >> 8) & 255;
  int c = i & 255;
  float v = (r >= c) ? Kk[layer * 256 + 255 - r + c] : 0.f;
  T[i] = (bf16)v;
}

// LayerNorm over D=128, one wave per row, bf16 output.
__global__ void ln_bf16_kernel(const float* __restrict__ X, const float* __restrict__ g,
                               const float* __restrict__ b, bf16* __restrict__ H) {
  int wave = threadIdx.x >> 5, lane = threadIdx.x & 31;
  size_t row = (size_t)blockIdx.x * 8 + wave;
  const float* x = X + row * DMOD;
  float v[4], s = 0.f;
#pragma unroll
  for (int j = 0; j < 4; ++j) { v[j] = x[lane * 4 + j]; s += v[j]; }
  float mean = wave_sum32(s) * (1.f / 128.f);
  float q = 0.f;
#pragma unroll
  for (int j = 0; j < 4; ++j) { float d = v[j] - mean; q += d * d; }
  float rstd = rsqrtf(wave_sum32(q) * (1.f / 128.f) + 1e-5f);
#pragma unroll
  for (int j = 0; j < 4; ++j) {
    int c = lane * 4 + j;
    H[row * DMOD + c] = (bf16)((v[j] - mean) * rstd * g[c] + b[c]);
  }
}

// Final LN: emits tokens (f32, to d_out) and L2-normalized tn (bf16).
__global__ void final_ln_kernel(const float* __restrict__ X, const float* __restrict__ g,
                                const float* __restrict__ b, float* __restrict__ tokens,
                                bf16* __restrict__ TN) {
  int wave = threadIdx.x >> 5, lane = threadIdx.x & 31;
  size_t row = (size_t)blockIdx.x * 8 + wave;
  const float* x = X + row * DMOD;
  float v[4], s = 0.f;
#pragma unroll
  for (int j = 0; j < 4; ++j) { v[j] = x[lane * 4 + j]; s += v[j]; }
  float mean = wave_sum32(s) * (1.f / 128.f);
  float q = 0.f;
#pragma unroll
  for (int j = 0; j < 4; ++j) { float d = v[j] - mean; q += d * d; }
  float rstd = rsqrtf(wave_sum32(q) * (1.f / 128.f) + 1e-5f);
  float t[4], sq = 0.f;
#pragma unroll
  for (int j = 0; j < 4; ++j) {
    int c = lane * 4 + j;
    t[j] = (v[j] - mean) * rstd * g[c] + b[c];
    tokens[row * DMOD + c] = t[j];
    sq += t[j] * t[j];
  }
  float inv = 1.f / fmaxf(sqrtf(wave_sum32(sq)), 1e-6f);
#pragma unroll
  for (int j = 0; j < 4; ++j) TN[row * DMOD + lane * 4 + j] = (bf16)(t[j] * inv);
}

// Depthwise causal conv K=4 -> bf16 in K-pair-interleaved layout per batch:
// xc16[b*65536 + (l>>1)*512 + 2c + (l&1)]
__global__ void dwconv_kernel(const float* __restrict__ xz, const float* __restrict__ cw,
                              const float* __restrict__ cb, bf16* __restrict__ xc16) {
  unsigned i = blockIdx.x * 256 + threadIdx.x;              // < MR*256
  int c = i & 255;
  int l = (i >> 8) & 255;
  long m = (long)(i >> 8);                                  // global row b*256 + l
  unsigned b = (unsigned)(m >> 8);
  float acc = cb[c];
#pragma unroll
  for (int k = 0; k < 4; ++k) {
    int li = l - 3 + k;
    if (li >= 0) acc += cw[c * 4 + k] * xz[(size_t)(m + (li - l)) * 512 + c];
  }
  size_t il = (size_t)b * 65536 + (size_t)(l >> 1) * 512 + (c << 1) + (l & 1);
  xc16[il] = (bf16)acc;
}

// y = (s4raw + D*xc) * sigmoid(z); xc read from interleaved layout, yg row-major.
__global__ void gating_kernel(const float* __restrict__ xz, const float* __restrict__ Dp,
                              const bf16* __restrict__ xc16, bf16* __restrict__ yg) {
  unsigned i = blockIdx.x * 256 + threadIdx.x;              // < MR*256
  int c = i & 255;
  size_t m = (size_t)(i >> 8);
  unsigned b = (unsigned)(m >> 8);
  int l = (int)(m & 255);
  size_t il = (size_t)b * 65536 + (size_t)(l >> 1) * 512 + (c << 1) + (l & 1);
  float s4v = xz[m * 512 + c];
  float zv  = xz[m * 512 + 256 + c];
  float y = (s4v + Dp[0] * (float)xc16[il]) * (1.f / (1.f + __expf(-zv)));
  yg[m * 256 + c] = (bf16)y;
}

// logits[m] = h1[m,:] . W2 + b2
__global__ void logits_kernel(const float* __restrict__ H1, const float* __restrict__ W2,
                              const float* __restrict__ b2, float* __restrict__ Lg) {
  size_t m = (size_t)blockIdx.x * 256 + threadIdx.x;        // < MR
  const float* h = H1 + m * 64;
  float acc = b2[0];
#pragma unroll
  for (int j = 0; j < 64; ++j) acc += h[j] * W2[j];
  Lg[m] = acc;
}

// Per-batch softmax over N=256 + weighted token pooling.
__global__ void softmax_feats_kernel(const float* __restrict__ Lg, const float* __restrict__ tokens,
                                     float* __restrict__ wOut, float* __restrict__ feats) {
  __shared__ float sw[256];
  __shared__ float red[256];
  int b = blockIdx.x, t = threadIdx.x;
  float lv = Lg[(size_t)b * 256 + t];
  red[t] = lv; __syncthreads();
  for (int s = 128; s; s >>= 1) { if (t < s) red[t] = fmaxf(red[t], red[t + s]); __syncthreads(); }
  float mx = red[0]; __syncthreads();
  float e = __expf(lv - mx);
  red[t] = e; __syncthreads();
  for (int s = 128; s; s >>= 1) { if (t < s) red[t] += red[t + s]; __syncthreads(); }
  float wv = e / red[0];
  sw[t] = wv;
  wOut[(size_t)b * 256 + t] = wv;
  __syncthreads();
  if (t < 128) {
    const float* tb = tokens + (size_t)b * 256 * 128 + t;
    float acc = 0.f;
    for (int n = 0; n < 256; ++n) acc += tb[(size_t)n * 128] * sw[n];
    feats[(size_t)b * 128 + t] = acc;
  }
}

// Three linear heads (64x128 each) off feats.
__global__ void heads_kernel(const float* __restrict__ feats,
                             const float* __restrict__ texW, const float* __restrict__ texb,
                             const float* __restrict__ edgW, const float* __restrict__ edgb,
                             const float* __restrict__ strW, const float* __restrict__ strb,
                             float* __restrict__ oTex, float* __restrict__ oEdg,
                             float* __restrict__ oStr) {
  int b = blockIdx.x, t = threadIdx.x;
  if (t >= 192) return;
  int head = t >> 6, j = t & 63;
  const float* W  = (head == 0) ? texW : (head == 1) ? edgW : strW;
  const float* bi = (head == 0) ? texb : (head == 1) ? edgb : strb;
  float* O        = (head == 0) ? oTex : (head == 1) ? oEdg : oStr;
  const float* f = feats + (size_t)b * 128;
  const float* w = W + (size_t)j * 128;
  float acc = bi[j];
#pragma unroll 8
  for (int k = 0; k < 128; ++k) acc += f[k] * w[k];
  O[(size_t)b * 64 + j] = acc;
}

// ---------------------------------------------------------------------------
// WMMA GEMM kernels: 8 waves/block, each wave owns a 16x64 output tile
// (4 accumulators -> A fragment reused 4x from registers).
// ---------------------------------------------------------------------------

// C[m*ldc + n] = A(MxK) @ W^T  (W is (N,K) row-major); block col span = 64.
__global__ void gemm_aw_store(const bf16* __restrict__ A, const bf16* __restrict__ W,
                              float* __restrict__ C, int K, int ldc) {
  int lane = threadIdx.x & 31, wave = threadIdx.x >> 5;
  size_t mBase = ((size_t)blockIdx.x * 8 + wave) * 16;
  int nBase = blockIdx.y * 64;
  f32x8 acc[4] = {};
  for (int kb = 0; kb < K; kb += 32) {
    bf16x16 a = load_a_frag(A, K, mBase, kb, lane);
#pragma unroll
    for (int t = 0; t < 4; ++t) {
      bf16x16 b = load_b_frag_wt(W, K, nBase + 16 * t, kb, lane);
      acc[t] = wmma_bf16(a, b, acc[t]);
    }
  }
  int half = lane >> 4, n0 = nBase + (lane & 15);
#pragma unroll
  for (int t = 0; t < 4; ++t)
#pragma unroll
    for (int r = 0; r < 8; ++r)
      C[(mBase + r + 8 * half) * (size_t)ldc + n0 + 16 * t] = acc[t][r];
}

// X = X + clip(rs,0,1.5) * (A @ W^T), X row-major ldc=128; block col span = 64.
__global__ void gemm_aw_res(const bf16* __restrict__ A, const bf16* __restrict__ W,
                            float* __restrict__ X, const float* __restrict__ rsp, int K) {
  int lane = threadIdx.x & 31, wave = threadIdx.x >> 5;
  size_t mBase = ((size_t)blockIdx.x * 8 + wave) * 16;
  int nBase = blockIdx.y * 64;
  f32x8 acc[4] = {};
  for (int kb = 0; kb < K; kb += 32) {
    bf16x16 a = load_a_frag(A, K, mBase, kb, lane);
#pragma unroll
    for (int t = 0; t < 4; ++t) {
      bf16x16 b = load_b_frag_wt(W, K, nBase + 16 * t, kb, lane);
      acc[t] = wmma_bf16(a, b, acc[t]);
    }
  }
  float rs = fminf(fmaxf(rsp[0], 0.f), 1.5f);
  int half = lane >> 4, n0 = nBase + (lane & 15);
#pragma unroll
  for (int t = 0; t < 4; ++t)
#pragma unroll
    for (int r = 0; r < 8; ++r) {
      size_t idx = (mBase + r + 8 * half) * (size_t)DMOD + n0 + 16 * t;
      X[idx] = X[idx] + rs * acc[t][r];
    }
}

// H1 = gelu(A @ W^T + bias); N=64 -> one block column group.
__global__ void gemm_aw_gelu(const bf16* __restrict__ A, const bf16* __restrict__ W,
                             const float* __restrict__ bias, float* __restrict__ C,
                             int K, int ldc) {
  int lane = threadIdx.x & 31, wave = threadIdx.x >> 5;
  size_t mBase = ((size_t)blockIdx.x * 8 + wave) * 16;
  f32x8 acc[4] = {};
  for (int kb = 0; kb < K; kb += 32) {
    bf16x16 a = load_a_frag(A, K, mBase, kb, lane);
#pragma unroll
    for (int t = 0; t < 4; ++t) {
      bf16x16 b = load_b_frag_wt(W, K, 16 * t, kb, lane);
      acc[t] = wmma_bf16(a, b, acc[t]);
    }
  }
  int half = lane >> 4, n0 = lane & 15;
#pragma unroll
  for (int t = 0; t < 4; ++t)
#pragma unroll
    for (int r = 0; r < 8; ++r) {
      int n = n0 + 16 * t;
      float v = acc[t][r] + bias[n];
      float g = 0.5f * v * (1.f + erff(v * 0.70710678118654752f));
      C[(mBase + r + 8 * half) * (size_t)ldc + n] = g;
    }
}

// Batched S4 long-conv GEMM: per batch, OUT(256x256) = T(256x256) @ Xc_b(256x256).
// Block: one 16-row tile of T staged into LDS (TDM), 8 waves x 32 channel cols.
// Result written into xz xp columns (ldc=512).
__global__ void gemm_s4(const bf16* __restrict__ T, const bf16* __restrict__ Xc,
                        float* __restrict__ C) {
  __shared__ bf16 shA[16 * 256];                            // 8 KB
  int lane = threadIdx.x & 31, wave = threadIdx.x >> 5;
  int mBase = blockIdx.x * 16;                              // token-row tile
  int bidx = blockIdx.y;                                    // batch
  const bf16* gsrc = T + (size_t)mBase * 256;

#if USE_TDM
  if (threadIdx.x == 0) {
    unsigned long long ga = (unsigned long long)gsrc;
    unsigned ldsAddr = (unsigned)(unsigned long long)(&shA[0]);
    u32x4 g0;
    g0[0] = 1u;                                             // count=1, no gather
    g0[1] = ldsAddr;                                        // lds_addr
    g0[2] = (unsigned)(ga & 0xffffffffu);                   // global_addr[31:0]
    g0[3] = ((unsigned)(ga >> 32) & 0x01ffffffu) | 0x80000000u; // ga[56:32] | type=2
    i32x8 g1;
    g1[0] = 0x00010000;                                     // data_size=1 (2 bytes)
    g1[1] = (int)(256u << 16);                              // tensor_dim0 = 256
    g1[2] = (int)(16u << 16);                               // tensor_dim1 = 16 (tile rows)
    g1[3] = (int)(256u << 16);                              // tile_dim0 = 256
    g1[4] = 16;                                             // tile_dim1 = 16, tile_dim2 = 0
    g1[5] = 256;                                            // tensor_dim0_stride = 256
    g1[6] = 0;
    g1[7] = 0;
    i32x4 gz4 = {0, 0, 0, 0};
    i32x8 gz8 = {0, 0, 0, 0, 0, 0, 0, 0};
    __builtin_amdgcn_tensor_load_to_lds(g0, g1, gz4, gz4, gz8, 0);
    __builtin_amdgcn_s_wait_tensorcnt(0);
  }
  __syncthreads();
#else
  {
    const unsigned* src = (const unsigned*)gsrc;
    unsigned* dst = (unsigned*)shA;
    for (int t = threadIdx.x; t < 16 * 256 / 2; t += 256) dst[t] = src[t];
    __syncthreads();
  }
#endif

  const bf16* Bm = Xc + (size_t)bidx * 65536;
  int nBase = wave * 32;
  f32x8 acc0 = {}, acc1 = {};
  for (int kb = 0; kb < 256; kb += 32) {
    bf16x16 a  = load_a_frag(shA, 256, 0, kb, lane);        // from LDS
    bf16x16 b0 = load_b_frag_il(Bm, nBase, kb, lane);
    bf16x16 b1 = load_b_frag_il(Bm, nBase + 16, kb, lane);
    acc0 = wmma_bf16(a, b0, acc0);
    acc1 = wmma_bf16(a, b1, acc1);
  }
  int half = lane >> 4, n0 = nBase + (lane & 15);
#pragma unroll
  for (int r = 0; r < 8; ++r) {
    size_t row = (size_t)bidx * 256 + mBase + r + 8 * half;
    C[row * 512 + n0]      = acc0[r];
    C[row * 512 + n0 + 16] = acc1[r];
  }
}

// ---------------------------------------------------------------------------
// Launch
// ---------------------------------------------------------------------------
extern "C" void kernel_launch(void* const* d_in, const int* in_sizes, int n_in,
                              void* d_out, int out_size, void* d_ws, size_t ws_size,
                              hipStream_t stream) {
  const float* tok0    = (const float*)d_in[0];
  const float* pos     = (const float*)d_in[1];
  const float* ln_g    = (const float*)d_in[2];
  const float* ln_b    = (const float*)d_in[3];
  const float* W_in    = (const float*)d_in[4];
  const float* conv_w  = (const float*)d_in[5];
  const float* conv_b  = (const float*)d_in[6];
  const float* ssm_B   = (const float*)d_in[7];
  const float* ssm_C   = (const float*)d_in[8];
  const float* log_dt  = (const float*)d_in[9];
  const float* ssm_D   = (const float*)d_in[10];
  const float* W_out   = (const float*)d_in[11];
  const float* res_sc  = (const float*)d_in[12];
  const float* fin_g   = (const float*)d_in[13];
  const float* fin_b   = (const float*)d_in[14];
  const float* imp_W1  = (const float*)d_in[15];
  const float* imp_b1  = (const float*)d_in[16];
  const float* imp_W2  = (const float*)d_in[17];
  const float* imp_b2  = (const float*)d_in[18];
  const float* tex_W   = (const float*)d_in[19];
  const float* tex_b   = (const float*)d_in[20];
  const float* edge_W  = (const float*)d_in[21];
  const float* edge_b  = (const float*)d_in[22];
  const float* str_W   = (const float*)d_in[23];
  const float* str_b   = (const float*)d_in[24];

  // Workspace layout
  char* ws = (char*)d_ws;
  size_t off = 0;
  float* bufX  = (float*)(ws + off); off += (size_t)MR * DMOD * 4;   // 64 MiB
  bf16*  bufH  = (bf16*) (ws + off); off += (size_t)MR * DMOD * 2;   // 32 MiB
  float* bufXZ = (float*)(ws + off); off += (size_t)MR * 512  * 4;   // 256 MiB
  bf16*  bufXC = (bf16*) (ws + off); off += (size_t)MR * DIN  * 2;   // 64 MiB (interleaved)
  bf16*  bufYG = (bf16*) (ws + off); off += (size_t)MR * DIN  * 2;   // 64 MiB (row-major)
  bf16*  Tmat  = (bf16*) (ws + off); off += (size_t)NLAY * 256 * 256 * 2;
  float* Kk    = (float*)(ws + off); off += (size_t)NLAY * 256 * 4;
  bf16*  Win16 = (bf16*) (ws + off); off += (size_t)NLAY * 512 * 128 * 2;
  bf16*  Wout16= (bf16*) (ws + off); off += (size_t)NLAY * 128 * 256 * 2;
  bf16*  W1_16 = (bf16*) (ws + off); off += (size_t)64 * 128 * 2;
  // bufXZ region is free after the layer loop: reuse for importance MLP.
  float* H1 = bufXZ;                              // MR x 64 f32
  float* Lg = bufXZ + (size_t)MR * 64;            // MR f32

  // Output layout (flat, return order)
  float* out_feats  = (float*)d_out;                                  // B*128
  float* out_tex    = out_feats  + (size_t)BB * DMOD;                 // B*64
  float* out_edge   = out_tex    + (size_t)BB * 64;
  float* out_str    = out_edge   + (size_t)BB * 64;
  float* out_tokens = out_str    + (size_t)BB * 64;                   // B*N*128
  float* out_w      = out_tokens + (size_t)MR * DMOD;                 // B*N

  dim3 blk(256);

  add_pos_kernel<<<(MR * DMOD) / 256, blk, 0, stream>>>(tok0, pos, bufX);
  cvt_bf16_kernel<<<(NLAY * 512 * 128 + 255) / 256, blk, 0, stream>>>(W_in, Win16, NLAY * 512 * 128);
  cvt_bf16_kernel<<<(NLAY * 128 * 256 + 255) / 256, blk, 0, stream>>>(W_out, Wout16, NLAY * 128 * 256);
  cvt_bf16_kernel<<<(64 * 128 + 255) / 256, blk, 0, stream>>>(imp_W1, W1_16, 64 * 128);
  kk_kernel<<<NLAY, 256, 0, stream>>>(ssm_B, ssm_C, log_dt, Kk);
  tbuild_kernel<<<(NLAY * 256 * 256) / 256, blk, 0, stream>>>(Kk, Tmat);

  for (int L = 0; L < NLAY; ++L) {
    ln_bf16_kernel<<<MR / 8, blk, 0, stream>>>(bufX, ln_g + L * DMOD, ln_b + L * DMOD, bufH);
    // xz = h @ W_in^T : (131072 x 512), K=128
    gemm_aw_store<<<dim3(MR / 16 / 8, 512 / 64), blk, 0, stream>>>(
        bufH, Win16 + (size_t)L * 512 * 128, bufXZ, 128, 512);
    dwconv_kernel<<<(MR * DIN) / 256, blk, 0, stream>>>(
        bufXZ, conv_w + (size_t)L * DIN * 4, conv_b + (size_t)L * DIN, bufXC);
    // s4raw = T @ xc per batch (TDM-staged T tile), written over xp columns of xz
    gemm_s4<<<dim3(16, BB), blk, 0, stream>>>(Tmat + (size_t)L * 256 * 256, bufXC, bufXZ);
    gating_kernel<<<(MR * DIN) / 256, blk, 0, stream>>>(bufXZ, ssm_D + L, bufXC, bufYG);
    // x += rs * (yg @ W_out^T) : (131072 x 128), K=256
    gemm_aw_res<<<dim3(MR / 16 / 8, DMOD / 64), blk, 0, stream>>>(
        bufYG, Wout16 + (size_t)L * 128 * 256, bufX, res_sc + L, 256);
  }

  final_ln_kernel<<<MR / 8, blk, 0, stream>>>(bufX, fin_g, fin_b, out_tokens, bufH);
  // h1 = gelu(tn @ imp_W1^T + b1) : (131072 x 64), K=128
  gemm_aw_gelu<<<MR / 16 / 8, blk, 0, stream>>>(bufH, W1_16, imp_b1, H1, 128, 64);
  logits_kernel<<<MR / 256, blk, 0, stream>>>(H1, imp_W2, imp_b2, Lg);
  softmax_feats_kernel<<<BB, blk, 0, stream>>>(Lg, out_tokens, out_w, out_feats);
  heads_kernel<<<BB, blk, 0, stream>>>(out_feats, tex_W, tex_b, edge_W, edge_b,
                                       str_W, str_b, out_tex, out_edge, out_str);
}
